// Network_22127671509083
// MI455X (gfx1250) — matrix operand order repaired and checked by reading, not compiled, not run on hardware
//
#include <hip/hip_runtime.h>

typedef __attribute__((ext_vector_type(16))) _Float16 v16h;
typedef __attribute__((ext_vector_type(8)))  _Float16 v8h;
typedef __attribute__((ext_vector_type(8)))  float    v8f;
typedef __attribute__((ext_vector_type(4)))  unsigned int v4u;
typedef __attribute__((ext_vector_type(8)))  int v8i;
typedef __attribute__((ext_vector_type(4)))  int v4i;

// Problem dims
#define BSZ   32
#define NIN   2312
#define NHID  512
#define NOUT  10
#define TS_N  350          // timesteps
#define KC    128          // K halves per LDS chunk (4 WMMA k-steps)
#define NCHUNK 19
#define K_P   (KC * NCHUNK)  // 2432: NIN padded so chunks divide evenly
#define T_P   384            // T padded to 24*16 (N tiles of 16, groups of 4)
#define LDS_ROW 136          // 128 data halves + 8 pad halves (68 dwords) per row

// Recurrence constants (exact closed forms)
#define D_SR  0.9048374180359595f   // exp(-Ts/tau_sr), tau_sr=10
#define C_SR  0.2718281828459045f   // Ts * e / tau_sr  (psp = C_SR * Bsr)
#define D_RF  0.3678794411714423f   // exp(-Ts/tau_ref), tau_ref=1
#define C_RF  -54.36563656918090f   // -scaleRef*theta*e*Ts/tau_ref = -2*10*e
#define THETA 10.0f

#if __has_builtin(__builtin_amdgcn_tensor_load_to_lds) && __has_builtin(__builtin_amdgcn_s_wait_tensorcnt)
#define USE_TDM 1
#else
#define USE_TDM 0
#endif

// ---------------- W1 fp32 -> f16, K padded to 2432 ----------------
__global__ void k_convert_w1(const float* __restrict__ W1, _Float16* __restrict__ W1h) {
  int idx = blockIdx.x * blockDim.x + threadIdx.x;
  if (idx >= NHID * K_P) return;
  int o = idx / K_P, i = idx % K_P;
  float v = (i < NIN) ? W1[o * NIN + i] : 0.0f;
  W1h[idx] = (_Float16)v;
}

// ------- spikeInput [b][i][t] fp32 -> Xh [b][t][i] f16, zero padded -------
__global__ void k_transpose_x(const float* __restrict__ X, _Float16* __restrict__ Xh) {
  __shared__ float tile[32][33];
  const int b  = blockIdx.z;
  const int i0 = blockIdx.x * 32;
  const int t0 = blockIdx.y * 32;
  const int tx = threadIdx.x, ty = threadIdx.y;   // 32 x 8
#pragma unroll
  for (int r = 0; r < 4; ++r) {
    int ii = ty + 8 * r;
    int gi = i0 + ii, gt = t0 + tx;
    float v = 0.0f;
    if (gi < NIN && gt < TS_N) v = X[((size_t)b * NIN + gi) * TS_N + gt];  // coalesced over t
    tile[ii][tx] = v;
  }
  __syncthreads();
#pragma unroll
  for (int r = 0; r < 4; ++r) {
    int tt = ty + 8 * r;
    Xh[((size_t)b * T_P + (t0 + tt)) * K_P + (i0 + tx)] = (_Float16)tile[tx][tt]; // coalesced over i
  }
}

// Fragment assembly helpers: concat two v8h into one v16h register block.
__device__ __forceinline__ v16h ld16(const _Float16* p) {          // 16 consecutive halves
  return __builtin_shufflevector(*(const v8h*)p, *(const v8h*)(p + 8),
                                 0,1,2,3,4,5,6,7,8,9,10,11,12,13,14,15);
}
__device__ __forceinline__ v16h ld16_a(const _Float16* p) {        // A frag: 8 halves, skip 8, 8 halves
  return __builtin_shufflevector(*(const v8h*)p, *(const v8h*)(p + 16),
                                 0,1,2,3,4,5,6,7,8,9,10,11,12,13,14,15);
}

#if USE_TDM
// Issue one TDM 2D tile load: 64 rows x 128 halves, global row stride K_P halves,
// LDS padding 4 dwords every 64 dwords (row stride 272B => every ds_load_b128
// hits each of the 64 banks exactly twice -- conflict-optimal).
// D# built per CDNA5 ISA 8.3/8.4. Wave-uniform; called from one wave per block.
// This toolchain's builtin takes 6 args: (g0 v4u, g1 v8i, g2 v4i, g3 v4i, v8i, cpol).
__device__ __forceinline__ void tdm_load_tile(const _Float16* gptr, unsigned lds_byte) {
  unsigned long long ga = (unsigned long long)(uintptr_t)gptr;
  v4u g0;
  g0[0] = 1u;                                       // count=1, user descriptor
  g0[1] = lds_byte;                                 // lds_addr
  g0[2] = (unsigned)(ga & 0xffffffffu);             // global_addr[31:0]
  g0[3] = (unsigned)((ga >> 32) & 0x1ffffffu) | (2u << 30);  // global_addr[56:32] | type=2
  v8i g1;
  g1[0] = (int)((1u << 16)      // data_size = 2 bytes
              | (1u << 20)      // pad_enable
              | (5u << 22)      // pad_interval: 64 dwords (256B = one tile row)
              | (3u << 25));    // pad_amount: 4 dwords (16B)
  g1[1] = (int)(((unsigned)K_P & 0xffffu) << 16);   // barrier_addr=0 | tensor_dim0 lo16
  g1[2] = (int)((((unsigned)K_P >> 16) & 0xffffu) | (64u << 16)); // dim0 hi | tensor_dim1 lo16
  g1[3] = (int)(128u << 16);                        // tensor_dim1 hi=0 | tile_dim0=128
  g1[4] = (int)(64u);                               // tile_dim1=64 | tile_dim2=0
  g1[5] = (int)K_P;                                 // tensor_dim0_stride[31:0]
  g1[6] = 0;                                        // stride hi16 | tensor_dim1_stride lo16
  g1[7] = 0;
  v4i z4 = {0, 0, 0, 0};                            // groups 2/3 unused (2D tensor)
  v8i z8 = {0, 0, 0, 0, 0, 0, 0, 0};                // extended dwords unused
  __builtin_amdgcn_tensor_load_to_lds(g0, g1, z4, z4, z8, 0);
}
#endif

// Cooperative fallback loader: same padded LDS layout as the TDM path.
__device__ __forceinline__ void coop_load_tile(const _Float16* __restrict__ gsrc_chunk,
                                               _Float16* __restrict__ buf, int tid) {
#pragma unroll
  for (int it = 0; it < 8; ++it) {
    int idx = tid + it * 128;       // 0..1023 : 64 rows x 16 segments of 8 halves
    int row = idx >> 4, seg = idx & 15;
    v8h v = *(const v8h*)(gsrc_chunk + (size_t)row * K_P + seg * 8);
    *(v8h*)(buf + row * LDS_ROW + seg * 8) = v;
  }
}

// ---------------- Layer-1 GEMM ----------------
// Block = 4 waves; each wave owns one 16(M) x 64(N) tile (4 WMMAs / k-step).
// B panel (64 t-rows x 128 K-halves) staged in LDS once per block via the
// Tensor Data Mover, double-buffered and overlapped with WMMA compute.
__global__ void __launch_bounds__(128)
k_gemm1(const _Float16* __restrict__ W1h, const _Float16* __restrict__ Xh,
        float* __restrict__ U1) {
  __shared__ _Float16 bbuf[2][64 * LDS_ROW];   // 2 x 17408 B

  const int tid = threadIdx.x;
  const unsigned lane = tid & 31u;
  const unsigned wave = tid >> 5;
  const unsigned hi   = lane >> 4;       // 0: lanes 0-15, 1: lanes 16-31
  const unsigned l16  = lane & 15u;
  const int b  = blockIdx.z;
  const int o0 = (blockIdx.x * 4 + wave) * 16;
  const int t0 = blockIdx.y * 64;

  // A frag: row o0+(lane&15); lanes 0-15 K {0..7,16..23}, lanes>=16 K {8..15,24..31}
  const _Float16* arow = W1h + (size_t)(o0 + l16) * K_P + hi * 8;
  const _Float16* gsrc = Xh + ((size_t)b * T_P + t0) * K_P;

  v8f acc0 = {}, acc1 = {}, acc2 = {}, acc3 = {};

#if USE_TDM
  if (wave == 0) {
    tdm_load_tile(gsrc, (unsigned)(uintptr_t)(&bbuf[0][0]));
    __builtin_amdgcn_s_wait_tensorcnt(0);
  }
  __syncthreads();
#else
  coop_load_tile(gsrc, &bbuf[0][0], tid);
  __syncthreads();
#endif

  for (int c = 0; c < NCHUNK; ++c) {
    const _Float16* bb = &bbuf[c & 1][0];
#if USE_TDM
    if (wave == 0 && c + 1 < NCHUNK)
      tdm_load_tile(gsrc + (size_t)(c + 1) * KC,
                    (unsigned)(uintptr_t)(&bbuf[(c + 1) & 1][0]));
#endif
#pragma unroll
    for (int ks = 0; ks < 4; ++ks) {
      v16h af = ld16_a(arow + c * KC + ks * 32);
      const int kb = ks * 32 + (int)(hi * 8);  // lanes 0-15: K {0..7,16..23}, lanes 16-31: K {8..15,24..31}
      v16h bf0 = ld16_a(bb + (0 * 16 + l16) * LDS_ROW + kb);
      v16h bf1 = ld16_a(bb + (1 * 16 + l16) * LDS_ROW + kb);
      v16h bf2 = ld16_a(bb + (2 * 16 + l16) * LDS_ROW + kb);
      v16h bf3 = ld16_a(bb + (3 * 16 + l16) * LDS_ROW + kb);
      acc0 = __builtin_amdgcn_wmma_f32_16x16x32_f16(false, af, false, bf0, (short)0, acc0, false, false);
      acc1 = __builtin_amdgcn_wmma_f32_16x16x32_f16(false, af, false, bf1, (short)0, acc1, false, false);
      acc2 = __builtin_amdgcn_wmma_f32_16x16x32_f16(false, af, false, bf2, (short)0, acc2, false, false);
      acc3 = __builtin_amdgcn_wmma_f32_16x16x32_f16(false, af, false, bf3, (short)0, acc3, false, false);
    }
#if USE_TDM
    if (wave == 0) __builtin_amdgcn_s_wait_tensorcnt(0);  // next buffer landed
    __syncthreads();  // + all waves done reading buf[c&1] before it is overwritten
#else
    __syncthreads();
    if (c + 1 < NCHUNK)
      coop_load_tile(gsrc + (size_t)(c + 1) * KC, &bbuf[(c + 1) & 1][0], tid);
    __syncthreads();
#endif
  }

  // C/D layout: VGPR r -> M = r + 8*hi, N = lane&15. Store U1[b][o][t].
  const int obase = o0 + (int)(hi * 8);
#pragma unroll
  for (int r = 0; r < 8; ++r) {
    float* urow = U1 + ((size_t)b * NHID + (obase + r)) * TS_N;
    int t;
    t = t0 +  0 + (int)l16; if (t < TS_N) urow[t] = acc0[r];
    t = t0 + 16 + (int)l16; if (t < TS_N) urow[t] = acc1[r];
    t = t0 + 32 + (int)l16; if (t < TS_N) urow[t] = acc2[r];
    t = t0 + 48 + (int)l16; if (t < TS_N) urow[t] = acc3[r];
  }
}

// ------- fused PSP (alpha kernel, 2-state) + refractory + threshold, layer 1 -------
__global__ void k_psp_spike1(const float* __restrict__ U1, float* __restrict__ S1) {
  int idx = blockIdx.x * blockDim.x + threadIdx.x;
  if (idx >= BSZ * NHID) return;
  const int b = idx / NHID, h = idx % NHID;
  const float* u = U1 + (size_t)idx * TS_N;
  float* sbase = S1 + (size_t)b * TS_N * NHID + h;
  float asr = 0.f, bsr = 0.f, x = 0.f, y = 0.f;
  for (int t = 0; t < TS_N; ++t) {
    if (t + 16 < TS_N) __builtin_prefetch(u + t + 16, 0, 0);
    bsr = D_SR * (bsr + asr);          // psp alpha-kernel state (lag>=1; k[0]=0)
    asr = D_SR * asr + u[t];
    float p = C_SR * bsr;
    y = D_RF * (y + x);                // refractory alpha-kernel state
    float um = p + C_RF * y;
    float spk = (um >= THETA) ? 1.0f : 0.0f;   // spike magnitude 1/Ts = 1
    x = D_RF * x + spk;
    sbase[(size_t)t * NHID] = spk;
  }
}

// ---------------- Layer-2 matmul: U2[b][o][t] = W2[o,:] . S1[b][t][:] ----------------
__global__ void k_fc2(const float* __restrict__ W2, const float* __restrict__ S1,
                      float* __restrict__ U2) {
  int idx = blockIdx.x * blockDim.x + threadIdx.x;
  if (idx >= BSZ * NOUT * TS_N) return;
  const int t = idx % TS_N;
  const int o = (idx / TS_N) % NOUT;
  const int b = idx / (TS_N * NOUT);
  const float* s = S1 + ((size_t)b * TS_N + t) * NHID;
  const float* w = W2 + (size_t)o * NHID;
  float acc = 0.f;
#pragma unroll 4
  for (int h = 0; h < NHID; h += 4) {
    float4 sv = *(const float4*)(s + h);
    float4 wv = *(const float4*)(w + h);
    acc = fmaf(sv.x, wv.x, acc);
    acc = fmaf(sv.y, wv.y, acc);
    acc = fmaf(sv.z, wv.z, acc);
    acc = fmaf(sv.w, wv.w, acc);
  }
  U2[idx] = acc;
}

// ------- fused PSP + refractory + threshold, layer 2 -> final output -------
__global__ void k_psp_spike2(const float* __restrict__ U2, float* __restrict__ out) {
  int idx = blockIdx.x * blockDim.x + threadIdx.x;
  if (idx >= BSZ * NOUT) return;
  const float* u = U2 + (size_t)idx * TS_N;
  float* so = out + (size_t)idx * TS_N;
  float asr = 0.f, bsr = 0.f, x = 0.f, y = 0.f;
  for (int t = 0; t < TS_N; ++t) {
    bsr = D_SR * (bsr + asr);
    asr = D_SR * asr + u[t];
    float p = C_SR * bsr;
    y = D_RF * (y + x);
    float um = p + C_RF * y;
    float spk = (um >= THETA) ? 1.0f : 0.0f;
    x = D_RF * x + spk;
    so[t] = spk;
  }
}

static inline size_t align256(size_t x) { return (x + 255) & ~(size_t)255; }

extern "C" void kernel_launch(void* const* d_in, const int* in_sizes, int n_in,
                              void* d_out, int out_size, void* d_ws, size_t ws_size,
                              hipStream_t stream) {
  (void)in_sizes; (void)n_in; (void)out_size; (void)ws_size;
  const float* X  = (const float*)d_in[0];   // (B, NIN, T) fp32
  const float* W1 = (const float*)d_in[1];   // (NHID, NIN) fp32
  const float* W2 = (const float*)d_in[2];   // (NOUT, NHID) fp32
  float* out = (float*)d_out;                // (B, NOUT, T) fp32

  char* ws = (char*)d_ws;
  size_t off = 0;
  _Float16* W1h = (_Float16*)(ws + off); off = align256(off + (size_t)NHID * K_P * sizeof(_Float16));
  _Float16* Xh  = (_Float16*)(ws + off); off = align256(off + (size_t)BSZ * T_P * K_P * sizeof(_Float16));
  float*    U1  = (float*)(ws + off);    off = align256(off + (size_t)BSZ * NHID * TS_N * sizeof(float));
  float*    S1  = (float*)(ws + off);    off = align256(off + (size_t)BSZ * TS_N * NHID * sizeof(float));
  float*    U2  = (float*)(ws + off);    off = align256(off + (size_t)BSZ * NOUT * TS_N * sizeof(float));

  { // 1) W1 -> f16 (K padded)
    int n = NHID * K_P;
    k_convert_w1<<<(n + 255) / 256, 256, 0, stream>>>(W1, W1h);
  }
  { // 2) transpose + convert input -> Xh[b][t][i] f16 (zero padded)
    dim3 grid(K_P / 32, T_P / 32, BSZ);
    dim3 block(32, 8);
    k_transpose_x<<<grid, block, 0, stream>>>(X, Xh);
  }
  { // 3) WMMA GEMM with TDM-staged LDS double buffering: U1[b][o][t]
    dim3 grid(NHID / 64, T_P / 64, BSZ);   // (8, 6, 32)
    k_gemm1<<<grid, 128, 0, stream>>>(W1h, Xh, U1);
  }
  { // 4) fused psp+spike layer 1 -> S1[b][t][h]
    int n = BSZ * NHID;
    k_psp_spike1<<<(n + 255) / 256, 256, 0, stream>>>(U1, S1);
  }
  { // 5) layer-2 matmul -> U2[b][o][t]
    int n = BSZ * NOUT * TS_N;
    k_fc2<<<(n + 255) / 256, 256, 0, stream>>>(W2, S1, U2);
  }
  { // 6) fused psp+spike layer 2 -> output
    int n = BSZ * NOUT;
    k_psp_spike2<<<(n + 255) / 256, 256, 0, stream>>>(U2, out);
  }
}
